// TemporalGraphAndGlobalFusion_16509854285880
// MI455X (gfx1250) — matrix operand order repaired and checked by reading, not compiled, not run on hardware
//
#include <hip/hip_runtime.h>
#include <hip/hip_bf16.h>

// Problem dimensions (fixed by the reference).
#define NN   131072   // nodes
#define NG   4096     // graphs
#define DZ   512
#define DX   256
#define DU   64
#define DH   512
#define DIN  768      // DZ + DX

// LDS B-tile staging: [2 buffers][512 cols][40 bf16]  (32 data + 8 pad) = 80 KB
#define COLSTRIDE 40
#define PARELEMS  (512 * COLSTRIDE)

typedef __attribute__((ext_vector_type(16))) __bf16 v16bf;
typedef __attribute__((ext_vector_type(8)))  __bf16 v8bf;
typedef __attribute__((ext_vector_type(8)))  float  v8f;

// ---- bf16 <-> f32 helpers (round-to-nearest-even, bit-level) ---------------
__device__ __forceinline__ __bf16 f2bf(float f) {
  union { float f; unsigned u; } in; in.f = f;
  unsigned u = in.u;
  unsigned r = u + 0x7FFFu + ((u >> 16) & 1u);
  union { unsigned short s; __bf16 b; } out; out.s = (unsigned short)(r >> 16);
  return out.b;
}
__device__ __forceinline__ float bf2f(__bf16 b) {
  union { __bf16 b; unsigned short s; } in; in.b = b;
  union { unsigned u; float f; } out; out.u = ((unsigned)in.s) << 16;
  return out.f;
}

// ---- WMMA A fragment (16x32 bf16), lane = M row (lane&15). ISA layout:
//   lanes 0-15 : K = kk+0..7   then kk+16..23
//   lanes 16-31: K = kk+8..15  then kk+24..31
__device__ __forceinline__ v16bf afrag_f32(const float* __restrict__ rowp, int kk, int lane) {
  const int b = kk + ((lane & 16) ? 8 : 0);
  const float4 f0 = *(const float4*)(rowp + b);
  const float4 f1 = *(const float4*)(rowp + b + 4);
  const float4 f2 = *(const float4*)(rowp + b + 16);
  const float4 f3 = *(const float4*)(rowp + b + 20);
  v16bf a;
  a[0]  = f2bf(f0.x); a[1]  = f2bf(f0.y); a[2]  = f2bf(f0.z); a[3]  = f2bf(f0.w);
  a[4]  = f2bf(f1.x); a[5]  = f2bf(f1.y); a[6]  = f2bf(f1.z); a[7]  = f2bf(f1.w);
  a[8]  = f2bf(f2.x); a[9]  = f2bf(f2.y); a[10] = f2bf(f2.z); a[11] = f2bf(f2.w);
  a[12] = f2bf(f3.x); a[13] = f2bf(f3.y); a[14] = f2bf(f3.z); a[15] = f2bf(f3.w);
  return a;
}

__device__ __forceinline__ v16bf afrag_bf16(const __bf16* __restrict__ rowp, int kk, int lane) {
  const int b = kk + ((lane & 16) ? 8 : 0);
  const v8bf lo = *(const v8bf*)(rowp + b);
  const v8bf hi = *(const v8bf*)(rowp + b + 16);
  v16bf a;
#pragma unroll
  for (int i = 0; i < 8; ++i) { a[i] = lo[i]; a[i + 8] = hi[i]; }
  return a;
}

#define WMMA_BF16(a, b, c) \
  __builtin_amdgcn_wmma_f32_16x16x32_bf16(false, (a), false, (b), (short)0, (c), false, false)

// ---- Async stage of one 32-K slice of W^T (all 512 cols) into LDS ----------
// Chunk c (16B): col = c>>2, quarter q = c&3. LDS col stride = 80B (padded).
// 2048 chunks / 512 threads = 4 async b128 copies per thread per k-step.
__device__ __forceinline__ void stage_async(const __bf16* __restrict__ WT, int K, int k0,
                                            unsigned ldsbase, int tid) {
#pragma unroll
  for (int i = 0; i < 4; ++i) {
    const int c   = tid + i * 512;
    const int col = c >> 2;
    const int q   = c & 3;
    const unsigned long long src =
        (unsigned long long)(uintptr_t)(WT + (size_t)col * K + k0 + q * 8);
    const unsigned dst = ldsbase + (unsigned)(col * (COLSTRIDE * 2) + q * 16);
    asm volatile("global_load_async_to_lds_b128 %0, %1, off"
                 :: "v"(dst), "v"(src) : "memory");
  }
}
#define WAIT_ASYNC_4() asm volatile("s_wait_asynccnt 0x4" ::: "memory")
#define WAIT_ASYNC_0() asm volatile("s_wait_asynccnt 0x0" ::: "memory")

// ---- Kernel 0: fp32 -> bf16 weight convert + transpose (WT[n*K+k] = W[k*N+n])
__global__ void __launch_bounds__(256) wcvt_kernel(const float* __restrict__ W,
                                                   __bf16* __restrict__ WT, int K, int N) {
  int idx = blockIdx.x * 256 + threadIdx.x;
  if (idx >= K * N) return;
  int k = idx / N;
  int n = idx - k * N;
  WT[(size_t)n * K + k] = f2bf(W[idx]);
}

// ---- Kernel 1: gates. R = sigmoid(X Wxr + h Whr + b), Z = sigmoid(X Wxz + h Whz + b)
// Writes P = h * R (bf16) and Z (bf16) to workspace.
// Workgroup: 64 rows x 512 cols. 16 waves = 4 M-tiles x 4 N-quarters (8 tiles/wave).
__global__ void __launch_bounds__(512) gru_gates_kernel(
    const float* __restrict__ z, const float* __restrict__ x,
    const float* __restrict__ prev_h,
    const __bf16* __restrict__ WxrT, const __bf16* __restrict__ WhrT,
    const __bf16* __restrict__ WxzT, const __bf16* __restrict__ WhzT,
    const float* __restrict__ bxr, const float* __restrict__ bhr,
    const float* __restrict__ bxz, const float* __restrict__ bhz,
    __bf16* __restrict__ Pws, __bf16* __restrict__ Zws) {
  __shared__ __align__(16) __bf16 Bbuf[2][PARELEMS];
  const int tid   = threadIdx.x;
  const int lane  = tid & 31;
  const int wave  = tid >> 5;
  const int rbase = blockIdx.x * 64 + (wave >> 2) * 16;
  const int ncol0 = (wave & 3) * 128;
  const int lrow  = rbase + (lane & 15);
  const float* zrow = z + (size_t)lrow * DZ;
  const float* xrow = x + (size_t)lrow * DX;
  const float* hrow = prev_h + (size_t)lrow * DH;
  const unsigned lds0 = (unsigned)(uintptr_t)&Bbuf[0][0];
  const int fragoff = (lane & 16) ? 8 : 0;

  for (int gate = 0; gate < 2; ++gate) {
    const __bf16* WxT = gate ? WxzT : WxrT;
    const __bf16* WhT = gate ? WhzT : WhrT;
    v8f acc[8] = {};

    // Double-buffered GEMM phase over one weight matrix.
    auto run_phase = [&](const __bf16* WT, int K, auto makeA) {
      stage_async(WT, K, 0, lds0, tid);                    // prologue -> buf 0
      int par = 0;
      for (int k0 = 0; k0 < K; k0 += 32) {
        const bool more = (k0 + 32) < K;
        if (more) stage_async(WT, K, k0 + 32, lds0 + (unsigned)((par ^ 1) * PARELEMS * 2), tid);
        const v16bf a = makeA(k0);                         // overlaps async wait
        if (more) { WAIT_ASYNC_4(); } else { WAIT_ASYNC_0(); }
        __syncthreads();                                   // everyone's stage visible
        const __bf16* bb = &Bbuf[par][0];
        auto ldb = [&](int t) {
          const int col = ncol0 + t * 16 + (lane & 15);
          const v8bf lo = *(const v8bf*)(bb + col * COLSTRIDE + fragoff);
          const v8bf hi = *(const v8bf*)(bb + col * COLSTRIDE + fragoff + 16);
          return __builtin_shufflevector(lo, hi, 0, 1, 2, 3, 4, 5, 6, 7, 8, 9, 10, 11, 12, 13, 14, 15);
        };
        // Software-pipelined B fragments: load t+1 while WMMA t executes.
        v16bf bcur = ldb(0);
#pragma unroll
        for (int t = 0; t < 8; ++t) {
          v16bf bnext = (t < 7) ? ldb(t + 1) : bcur;
          acc[t] = WMMA_BF16(a, bcur, acc[t]);
          bcur = bnext;
        }
        __syncthreads();                                   // done reading buf[par]
        par ^= 1;
      }
    };

    // X @ Wx  (K=768; 32-steps never straddle the z|x boundary: 512 % 32 == 0)
    run_phase(WxT, DIN, [&](int k0) {
      const float* rp; int kk;
      if (k0 < DZ) { rp = zrow; kk = k0; } else { rp = xrow; kk = k0 - DZ; }
      return afrag_f32(rp, kk, lane);
    });
    // prev_h @ Wh  (K=512)
    run_phase(WhT, DH, [&](int k0) { return afrag_f32(hrow, k0, lane); });

    // Epilogue. C tile: VGPR j -> lanes 0-15: (M=j, N=lane), lanes 16-31: (M=j+8)
    const float* bx = gate ? bxz : bxr;
    const float* bh = gate ? bhz : bhr;
    __bf16* dst = gate ? Zws : Pws;
#pragma unroll
    for (int t = 0; t < 8; ++t) {
      const int col = ncol0 + t * 16 + (lane & 15);
      const float bias = bx[col] + bh[col];
#pragma unroll
      for (int j = 0; j < 8; ++j) {
        const int row = rbase + j + ((lane & 16) ? 8 : 0);
        const size_t idx = (size_t)row * DH + col;
        const float s = 1.0f / (1.0f + __expf(-(acc[t][j] + bias)));
        const float val = gate ? s : s * prev_h[idx];      // Z itself, or P = h*R
        dst[idx] = f2bf(val);
      }
    }
  }
}

// ---- Kernel 2: candidate + combine. H = Z*h + (1-Z)*tanh(X Wxh + P Whh + b)
__global__ void __launch_bounds__(512) gru_cand_kernel(
    const float* __restrict__ z, const float* __restrict__ x,
    const float* __restrict__ prev_h,
    const __bf16* __restrict__ Pws, const __bf16* __restrict__ Zws,
    const __bf16* __restrict__ WxhT, const __bf16* __restrict__ WhhT,
    const float* __restrict__ bxh, const float* __restrict__ bhh,
    float* __restrict__ Hout) {
  __shared__ __align__(16) __bf16 Bbuf[2][PARELEMS];
  const int tid   = threadIdx.x;
  const int lane  = tid & 31;
  const int wave  = tid >> 5;
  const int rbase = blockIdx.x * 64 + (wave >> 2) * 16;
  const int ncol0 = (wave & 3) * 128;
  const int lrow  = rbase + (lane & 15);
  const float*  zrow = z + (size_t)lrow * DZ;
  const float*  xrow = x + (size_t)lrow * DX;
  const __bf16* prow = Pws + (size_t)lrow * DH;
  const unsigned lds0 = (unsigned)(uintptr_t)&Bbuf[0][0];
  const int fragoff = (lane & 16) ? 8 : 0;

  v8f acc[8] = {};

  auto run_phase = [&](const __bf16* WT, int K, auto makeA) {
    stage_async(WT, K, 0, lds0, tid);
    int par = 0;
    for (int k0 = 0; k0 < K; k0 += 32) {
      const bool more = (k0 + 32) < K;
      if (more) stage_async(WT, K, k0 + 32, lds0 + (unsigned)((par ^ 1) * PARELEMS * 2), tid);
      const v16bf a = makeA(k0);
      if (more) { WAIT_ASYNC_4(); } else { WAIT_ASYNC_0(); }
      __syncthreads();
      const __bf16* bb = &Bbuf[par][0];
      auto ldb = [&](int t) {
        const int col = ncol0 + t * 16 + (lane & 15);
        const v8bf lo = *(const v8bf*)(bb + col * COLSTRIDE + fragoff);
        const v8bf hi = *(const v8bf*)(bb + col * COLSTRIDE + fragoff + 16);
        return __builtin_shufflevector(lo, hi, 0, 1, 2, 3, 4, 5, 6, 7, 8, 9, 10, 11, 12, 13, 14, 15);
      };
      v16bf bcur = ldb(0);
#pragma unroll
      for (int t = 0; t < 8; ++t) {
        v16bf bnext = (t < 7) ? ldb(t + 1) : bcur;
        acc[t] = WMMA_BF16(a, bcur, acc[t]);
        bcur = bnext;
      }
      __syncthreads();
      par ^= 1;
    }
  };

  run_phase(WxhT, DIN, [&](int k0) {
    const float* rp; int kk;
    if (k0 < DZ) { rp = zrow; kk = k0; } else { rp = xrow; kk = k0 - DZ; }
    return afrag_f32(rp, kk, lane);
  });
  run_phase(WhhT, DH, [&](int k0) { return afrag_bf16(prow, k0, lane); });

#pragma unroll
  for (int t = 0; t < 8; ++t) {
    const int col = ncol0 + t * 16 + (lane & 15);
    const float bias = bxh[col] + bhh[col];
#pragma unroll
    for (int j = 0; j < 8; ++j) {
      const int row = rbase + j + ((lane & 16) ? 8 : 0);
      const size_t idx = (size_t)row * DH + col;
      const float ht = tanhf(acc[t][j] + bias);
      const float zg = bf2f(Zws[idx]);
      Hout[idx] = zg * prev_h[idx] + (1.0f - zg) * ht;
    }
  }
}

// ---- Kernel 3: segment_sum(z, batch) — batch sorted; one block per graph,
// binary-searched row range, no atomics, deterministic.
__global__ void __launch_bounds__(256) segsum_kernel(const float* __restrict__ z,
                                                     const long long* __restrict__ batch,
                                                     float* __restrict__ out) {
  const long long g = (long long)blockIdx.x;
  int lo = 0, hi = NN;
  while (lo < hi) { int m = (lo + hi) >> 1; if (batch[m] < g) lo = m + 1; else hi = m; }
  const int start = lo;
  hi = NN;
  while (lo < hi) { int m = (lo + hi) >> 1; if (batch[m] < g + 1) lo = m + 1; else hi = m; }
  const int end = lo;
  const int c = threadIdx.x * 2;
  float s0 = 0.0f, s1 = 0.0f;
  for (int r = start; r < end; ++r) {
    const float2 v = *(const float2*)(z + (size_t)r * DZ + c);
    s0 += v.x; s1 += v.y;
  }
  float* o = out + (size_t)blockIdx.x * 1024 + c;   // fused row stride = 2*DZ
  o[0] = s0; o[1] = s1;
}

// ---- Kernel 4: glob_emb = relu(u @ Wg + bg)  (tiny: 268 MFLOP, plain fp32)
__global__ void __launch_bounds__(256) globproj_kernel(const float* __restrict__ u,
                                                       const float* __restrict__ Wg,
                                                       const float* __restrict__ bg,
                                                       float* __restrict__ out) {
  __shared__ float su[DU];
  const int g = blockIdx.x;
  if (threadIdx.x < DU) su[threadIdx.x] = u[(size_t)g * DU + threadIdx.x];
  __syncthreads();
  const int c = threadIdx.x * 2;
  float s0 = bg[c], s1 = bg[c + 1];
#pragma unroll 8
  for (int k = 0; k < DU; ++k) {
    const float2 w = *(const float2*)(Wg + (size_t)k * DZ + c);
    s0 += su[k] * w.x; s1 += su[k] * w.y;
  }
  s0 = s0 > 0.0f ? s0 : 0.0f;
  s1 = s1 > 0.0f ? s1 : 0.0f;
  float* o = out + (size_t)g * 1024 + DZ + c;
  o[0] = s0; o[1] = s1;
}

extern "C" void kernel_launch(void* const* d_in, const int* in_sizes, int n_in,
                              void* d_out, int out_size, void* d_ws, size_t ws_size,
                              hipStream_t stream) {
  const float* z      = (const float*)d_in[0];
  const float* u      = (const float*)d_in[1];
  const float* x      = (const float*)d_in[2];
  const float* prev_h = (const float*)d_in[3];
  const long long* batch = (const long long*)d_in[5];   // int64, sorted
  const float* Wxz = (const float*)d_in[7];  const float* bxz = (const float*)d_in[8];
  const float* Whz = (const float*)d_in[9];  const float* bhz = (const float*)d_in[10];
  const float* Wxr = (const float*)d_in[11]; const float* bxr = (const float*)d_in[12];
  const float* Whr = (const float*)d_in[13]; const float* bhr = (const float*)d_in[14];
  const float* Wxh = (const float*)d_in[15]; const float* bxh = (const float*)d_in[16];
  const float* Whh = (const float*)d_in[17]; const float* bhh = (const float*)d_in[18];
  const float* Wg  = (const float*)d_in[19]; const float* bg  = (const float*)d_in[20];

  // Workspace carve-out (deterministic, same every call).
  char* wsp = (char*)d_ws;
  size_t off = 0;
  auto carve = [&](size_t bytes) -> void* {
    void* p = wsp + off;
    off += (bytes + 255) & ~(size_t)255;
    return p;
  };
  __bf16* WxrT = (__bf16*)carve((size_t)DIN * DH * 2);
  __bf16* WxzT = (__bf16*)carve((size_t)DIN * DH * 2);
  __bf16* WxhT = (__bf16*)carve((size_t)DIN * DH * 2);
  __bf16* WhrT = (__bf16*)carve((size_t)DH * DH * 2);
  __bf16* WhzT = (__bf16*)carve((size_t)DH * DH * 2);
  __bf16* WhhT = (__bf16*)carve((size_t)DH * DH * 2);
  __bf16* Pws  = (__bf16*)carve((size_t)NN * DH * 2);
  __bf16* Zws  = (__bf16*)carve((size_t)NN * DH * 2);

  // 0) Weight convert + transpose (bf16 W^T, ~4MB total — stays in L2)
  wcvt_kernel<<<(DIN * DH + 255) / 256, 256, 0, stream>>>(Wxr, WxrT, DIN, DH);
  wcvt_kernel<<<(DIN * DH + 255) / 256, 256, 0, stream>>>(Wxz, WxzT, DIN, DH);
  wcvt_kernel<<<(DIN * DH + 255) / 256, 256, 0, stream>>>(Wxh, WxhT, DIN, DH);
  wcvt_kernel<<<(DH * DH + 255) / 256, 256, 0, stream>>>(Whr, WhrT, DH, DH);
  wcvt_kernel<<<(DH * DH + 255) / 256, 256, 0, stream>>>(Whz, WhzT, DH, DH);
  wcvt_kernel<<<(DH * DH + 255) / 256, 256, 0, stream>>>(Whh, WhhT, DH, DH);

  // 1) R/Z gates -> P, Z in workspace
  gru_gates_kernel<<<NN / 64, 512, 0, stream>>>(z, x, prev_h,
                                                WxrT, WhrT, WxzT, WhzT,
                                                bxr, bhr, bxz, bhz, Pws, Zws);
  // 2) Candidate + GRU combine -> H (second half of d_out)
  float* Hout = (float*)d_out + (size_t)NG * 1024;
  gru_cand_kernel<<<NN / 64, 512, 0, stream>>>(z, x, prev_h, Pws, Zws,
                                               WxhT, WhhT, bxh, bhh, Hout);
  // 3) fused_emb[:, :512] = segment_sum(z, batch)
  segsum_kernel<<<NG, 256, 0, stream>>>(z, batch, (float*)d_out);
  // 4) fused_emb[:, 512:] = relu(u @ Wg + bg)
  globproj_kernel<<<NG, 256, 0, stream>>>(u, Wg, bg, (float*)d_out);
}
